// DeRAAttention_45389214384190
// MI455X (gfx1250) — matrix operand order repaired and checked by reading, not compiled, not run on hardware
//
#include <hip/hip_runtime.h>
#include <hip/hip_bf16.h>

// ---------------------------------------------------------------------------
// DeRA spatial-RoPE attention for MI455X (gfx1250), bf16 WMMA + fp32 accum.
// B=1, S=3072 (48x64 grid), DIM=1536, NH=16, HD=96.
// Round 2: async global->LDS staging (ASYNCcnt) + double-buffered tiles.
// ---------------------------------------------------------------------------

#define S_LEN   3072
#define DIMSZ   1536
#define NHEADS  16
#define HDIM    96
#define GRID_H  48
#define GRID_W  64
#define CROPE   48          // HD/2
#define C3      16          // CROPE/3
#define BKG     64          // GEMM K-tile

typedef __bf16 bf16_t;
typedef bf16_t v16bf  __attribute__((ext_vector_type(16)));
typedef bf16_t bf16x8 __attribute__((ext_vector_type(8)));
typedef float  v8f    __attribute__((ext_vector_type(8)));

__device__ __forceinline__ bf16_t f2bf(float f) {
    union { float f; unsigned u; } a; a.f = f;
    unsigned r = (a.u + 0x7FFFu + ((a.u >> 16) & 1u)) >> 16;
    unsigned short s = (unsigned short)r;
    union { unsigned short s; bf16_t b; } o; o.s = s;
    return o.b;
}

__device__ __forceinline__ v16bf pack16(bf16x8 a, bf16x8 b) {
    return __builtin_shufflevector(a, b, 0, 1, 2, 3, 4, 5, 6, 7,
                                          8, 9, 10, 11, 12, 13, 14, 15);
}

__device__ __forceinline__ v8f wmma_bf16(v16bf a, v16bf b, v8f c) {
    // D(f32 16x16) = A(bf16 16x32) * B(bf16 32x16) + C
    return __builtin_amdgcn_wmma_f32_16x16x32_bf16(
        /*neg_a=*/false, a, /*neg_b=*/false, b,
        /*c_mod=*/(short)0, c, /*reuse_a=*/false, /*reuse_b=*/false);
}

// Async DMA: 16 bytes per lane, global -> LDS, tracked by ASYNCcnt.
// GVS addressing: mem = SADDR(64b SGPR) + VADDR(32b VGPR byte offset).
// vdst = per-lane LDS byte address (generic LDS pointers carry the LDS
// offset in addr[31:0]).
__device__ __forceinline__ void async_b128(const void* sbase, unsigned gbyteoff,
                                           void* lds_ptr) {
    unsigned lds = (unsigned)(reinterpret_cast<uintptr_t>(lds_ptr));
    asm volatile("global_load_async_to_lds_b128 %0, %1, %2"
                 :: "v"(lds), "v"(gbyteoff), "s"(sbase)
                 : "memory");
}
__device__ __forceinline__ void wait_async0() {
    asm volatile("s_wait_asynccnt 0" ::: "memory");
}

// ---------------------------------------------------------------------------
// Kernel 1a: fp32 -> bf16 straight copy (activation x)
// ---------------------------------------------------------------------------
__global__ void cvt_x_kernel(const float* __restrict__ x,
                             bf16_t* __restrict__ xbf, int n) {
    int t = blockIdx.x * 256 + threadIdx.x;
    if (t < n) xbf[t] = f2bf(x[t]);
}

// ---------------------------------------------------------------------------
// Kernel 1b: fp32 W[k][n] -> bf16 Wt[n][k] (transposed), 4 weights via grid.z
// ---------------------------------------------------------------------------
__global__ void cvt_wt_kernel(const float* __restrict__ w0, const float* __restrict__ w1,
                              const float* __restrict__ w2, const float* __restrict__ w3,
                              bf16_t* __restrict__ o0, bf16_t* __restrict__ o1,
                              bf16_t* __restrict__ o2, bf16_t* __restrict__ o3) {
    const float* w; bf16_t* o;
    switch (blockIdx.z) {
        case 0:  w = w0; o = o0; break;
        case 1:  w = w1; o = o1; break;
        case 2:  w = w2; o = o2; break;
        default: w = w3; o = o3; break;
    }
    int t = blockIdx.x * 256 + threadIdx.x;      // over DIM*DIM
    int k = t / DIMSZ, n = t % DIMSZ;
    o[(size_t)n * DIMSZ + k] = f2bf(w[t]);
}

// ---------------------------------------------------------------------------
// Kernel 2: C[m][n] = sum_k A[m][k]*Wt[n][k] + bias[n]
// M=3072, N=1536, K=1536. 256 threads = 8 waves, block tile 128x128, BK=64,
// double-buffered LDS fed by async global->LDS DMA. Wave grid 2(m) x 4(n);
// each wave owns 64x32 = 4x2 WMMA tiles, 16 WMMA per K-step.
// ---------------------------------------------------------------------------
__global__ __launch_bounds__(256) void gemm_bf16_kernel(
    const bf16_t* __restrict__ A,    // [3072][1536] row-major bf16
    const bf16_t* __restrict__ Wt,   // [1536][1536] row-major bf16 (W^T)
    const float*  __restrict__ bias, // [1536]
    float*        __restrict__ C) {  // [3072][1536] fp32
    __shared__ __align__(16) bf16_t Asmem[2][128 * BKG];   // 16 KB x2
    __shared__ __align__(16) bf16_t Bsmem[2][128 * BKG];   // 16 KB x2

    const int tid  = threadIdx.x;
    const int wave = tid >> 5, lane = tid & 31;
    const int lo = lane & 15, hi = lane >> 4;
    const int wm = wave >> 2, wn = wave & 3;
    const int m0 = blockIdx.y * 128, n0 = blockIdx.x * 128;

    // stage one 128xBKG tile pair into buffer b via async DMA (8 instr/wave)
    auto stage = [&](int k0, int b) {
#pragma unroll
        for (int c = 0; c < 4; ++c) {
            int ci  = tid + c * 256;         // 1024 chunks of 8 bf16
            int row = ci >> 3;               // 8 chunks per row
            int col = (ci & 7) * 8;
            async_b128(A,  (unsigned)(((m0 + row) * DIMSZ + k0 + col) * 2),
                       &Asmem[b][row * BKG + col]);
            async_b128(Wt, (unsigned)(((n0 + row) * DIMSZ + k0 + col) * 2),
                       &Bsmem[b][row * BKG + col]);
        }
    };

    v8f acc[4][2];
#pragma unroll
    for (int i = 0; i < 4; ++i)
#pragma unroll
        for (int j = 0; j < 2; ++j) acc[i][j] = (v8f)(0.0f);

    stage(0, 0);
    const int NIT = DIMSZ / BKG;             // 24
    for (int it = 0; it < NIT; ++it) {
        const int b = it & 1;
        wait_async0();                       // my DMA for buffer b done
        __syncthreads();                     // everyone's DMA done; prev reads done
        if (it + 1 < NIT) stage((it + 1) * BKG, b ^ 1);  // prefetch under compute

#pragma unroll
        for (int kc = 0; kc < 2; ++kc) {     // two 32-wide K chunks
            v16bf af[4];
#pragma unroll
            for (int i = 0; i < 4; ++i) {
                int row = wm * 64 + i * 16 + lo;          // A: lane -> M row
                bf16x8 a0 = *(const bf16x8*)&Asmem[b][row * BKG + kc * 32 + hi * 8];
                bf16x8 a1 = *(const bf16x8*)&Asmem[b][row * BKG + kc * 32 + 16 + hi * 8];
                af[i] = pack16(a0, a1);
            }
            v16bf bfv[2];
#pragma unroll
            for (int j = 0; j < 2; ++j) {
                int row = wn * 32 + j * 16 + lo;          // B: lane -> N col
                bf16x8 b0 = *(const bf16x8*)&Bsmem[b][row * BKG + kc * 32 + hi * 8];
                bf16x8 b1 = *(const bf16x8*)&Bsmem[b][row * BKG + kc * 32 + 16 + hi * 8];
                bfv[j] = pack16(b0, b1);
            }
#pragma unroll
            for (int i = 0; i < 4; ++i)
#pragma unroll
                for (int j = 0; j < 2; ++j)
                    acc[i][j] = wmma_bf16(af[i], bfv[j], acc[i][j]);
        }
    }

#pragma unroll
    for (int i = 0; i < 4; ++i)
#pragma unroll
        for (int j = 0; j < 2; ++j)
#pragma unroll
            for (int r = 0; r < 8; ++r) {
                int m = m0 + wm * 64 + i * 16 + hi * 8 + r;   // C: M = hi*8+r
                int n = n0 + wn * 32 + j * 16 + lo;           // C: N = lo
                C[(size_t)m * DIMSZ + n] = acc[i][j][r] + bias[n];
            }
}

// ---------------------------------------------------------------------------
// Kernel 3: spatial RoPE on Q/K, pack Q,K,V to bf16.
// qkv fp32 [3][S][DIM]; outputs Qbf/Kbf [NH][S][HD] (Q pre-scaled by 1/sqrt(96)),
// Vt [NH][HD][S] (transposed so flash B-fragments are contiguous).
// ---------------------------------------------------------------------------
__global__ void rope_pack_kernel(const float* __restrict__ qkv,
                                 const float* __restrict__ fcos,
                                 const float* __restrict__ fsin,
                                 bf16_t* __restrict__ Qbf,
                                 bf16_t* __restrict__ Kbf,
                                 bf16_t* __restrict__ Vt) {
    int t = blockIdx.x * 256 + threadIdx.x;          // over S*NH*CROPE
    if (t >= S_LEN * NHEADS * CROPE) return;
    int c = t % CROPE;
    int rem = t / CROPE;
    int h = rem % NHEADS;
    int s = rem / NHEADS;
    int d = 2 * c;

    size_t base = (size_t)s * DIMSZ + h * HDIM + d;
    float qr = qkv[base], qi = qkv[base + 1];
    float kr = qkv[(size_t)S_LEN * DIMSZ + base];
    float ki = qkv[(size_t)S_LEN * DIMSZ + base + 1];
    float vr = qkv[2 * (size_t)S_LEN * DIMSZ + base];
    float vi = qkv[2 * (size_t)S_LEN * DIMSZ + base + 1];

    float cs = 1.0f, sn = 0.0f;
    if (c >= C3) {
        int p = (c < 2 * C3) ? (s / GRID_W) : (s % GRID_W);
        cs = fcos[p * CROPE + c];
        sn = fsin[p * CROPE + c];
    }
    const float scale = 0.10206207261596577f;        // 1/sqrt(96)
    float qyr = (qr * cs - qi * sn) * scale;
    float qyi = (qr * sn + qi * cs) * scale;
    float kyr = kr * cs - ki * sn;
    float kyi = kr * sn + ki * cs;

    size_t ob = ((size_t)h * S_LEN + s) * HDIM + d;
    Qbf[ob]     = f2bf(qyr);
    Qbf[ob + 1] = f2bf(qyi);
    Kbf[ob]     = f2bf(kyr);
    Kbf[ob + 1] = f2bf(kyi);
    Vt[((size_t)h * HDIM + d)     * S_LEN + s] = f2bf(vr);
    Vt[((size_t)h * HDIM + d + 1) * S_LEN + s] = f2bf(vi);
}

// ---------------------------------------------------------------------------
// Kernel 4: flash attention. grid = (S/128, NH); 256 threads = 8 waves, each
// wave owns 16 q-rows x 96 d. Streams 64-wide K/V tiles through
// double-buffered LDS fed by async DMA.
// ---------------------------------------------------------------------------
__global__ __launch_bounds__(256) void flash_kernel(
    const bf16_t* __restrict__ Q,    // [NH][S][HD], pre-scaled
    const bf16_t* __restrict__ K,    // [NH][S][HD]
    const bf16_t* __restrict__ Vt,   // [NH][HD][S]
    bf16_t*       __restrict__ O) {  // [S][DIM] bf16
    __shared__ __align__(16) bf16_t Ks[2][64 * 96];     // [kcol][hd]   12 KB x2
    __shared__ __align__(16) bf16_t Vs[2][96 * 64];     // [d][kcol]    12 KB x2
    __shared__ __align__(16) bf16_t Ps[8 * 16 * 64];    // per-wave P   16 KB

    const int tid  = threadIdx.x;
    const int wave = tid >> 5, lane = tid & 31;
    const int lo = lane & 15, hi = lane >> 4;
    const int head = blockIdx.y;
    const int q0   = blockIdx.x * 128;
    const int qg   = q0 + wave * 16 + lo;

    // stage K tile [64][96] and V tile [96][64] into buffer b (6 instr/wave)
    auto stageKV = [&](int k0, int b) {
#pragma unroll
        for (int c = 0; c < 3; ++c) {
            int ci = tid + c * 256;                      // 768 chunks each
            int krow = ci / 12, kcol = (ci % 12) * 8;
            async_b128(K, (unsigned)((((head * S_LEN) + k0 + krow) * HDIM + kcol) * 2),
                       &Ks[b][krow * 96 + kcol]);
            int vrow = ci >> 3, vcol = (ci & 7) * 8;
            async_b128(Vt, (unsigned)((((head * HDIM) + vrow) * S_LEN + k0 + vcol) * 2),
                       &Vs[b][vrow * 64 + vcol]);
        }
    };

    // Q fragments: 3 K-chunks of 32 over HD=96, held in registers for the run
    v16bf qf[3];
    {
        const bf16_t* qrow = Q + ((size_t)head * S_LEN + qg) * HDIM;
#pragma unroll
        for (int c = 0; c < 3; ++c) {
            bf16x8 a0 = *(const bf16x8*)(qrow + 32 * c + hi * 8);
            bf16x8 a1 = *(const bf16x8*)(qrow + 32 * c + 16 + hi * 8);
            qf[c] = pack16(a0, a1);
        }
    }

    v8f o_acc[6];
#pragma unroll
    for (int dn = 0; dn < 6; ++dn) o_acc[dn] = (v8f)(0.0f);
    float m_run[8], l_run[8];
#pragma unroll
    for (int r = 0; r < 8; ++r) { m_run[r] = -1e30f; l_run[r] = 0.0f; }

    stageKV(0, 0);
    const int NIT = S_LEN / 64;                          // 48
    for (int it = 0; it < NIT; ++it) {
        const int b = it & 1;
        wait_async0();
        __syncthreads();
        if (it + 1 < NIT) stageKV((it + 1) * 64, b ^ 1); // prefetch under compute

        // S = Q * K^T : 4 N-tiles of 16 k-cols, 3 WMMA each over HD
        v8f sc[4];
#pragma unroll
        for (int nt = 0; nt < 4; ++nt) {
            sc[nt] = (v8f)(0.0f);
#pragma unroll
            for (int c = 0; c < 3; ++c) {
                int krow = nt * 16 + lo;                 // B: lane -> k col
                bf16x8 b0 = *(const bf16x8*)&Ks[b][krow * 96 + c * 32 + hi * 8];
                bf16x8 b1 = *(const bf16x8*)&Ks[b][krow * 96 + c * 32 + 16 + hi * 8];
                sc[nt] = wmma_bf16(qf[c], pack16(b0, b1), sc[nt]);
            }
        }

        // online softmax per row (row m = hi*8 + r, cols spread over 16 lanes)
        float corr[8];
#pragma unroll
        for (int r = 0; r < 8; ++r) {
            float mx = fmaxf(fmaxf(sc[0][r], sc[1][r]), fmaxf(sc[2][r], sc[3][r]));
#pragma unroll
            for (int off = 1; off < 16; off <<= 1)
                mx = fmaxf(mx, __shfl_xor(mx, off, 32));
            float mnew = fmaxf(m_run[r], mx);
            corr[r] = __expf(m_run[r] - mnew);
            m_run[r] = mnew;
            float psum = 0.0f;
#pragma unroll
            for (int nt = 0; nt < 4; ++nt) {
                float p = __expf(sc[nt][r] - mnew);
                psum += p;
                Ps[wave * 1024 + (hi * 8 + r) * 64 + nt * 16 + lo] = f2bf(p);
            }
#pragma unroll
            for (int off = 1; off < 16; off <<= 1)
                psum += __shfl_xor(psum, off, 32);
            l_run[r] = l_run[r] * corr[r] + psum;
        }
        asm volatile("s_wait_dscnt 0" ::: "memory");   // P LDS writes visible in-wave

#pragma unroll
        for (int dn = 0; dn < 6; ++dn)
#pragma unroll
            for (int r = 0; r < 8; ++r) o_acc[dn][r] *= corr[r];

        // O += P * V : 2 K-chunks of 32 over the 64 k-cols, 6 d-tiles
#pragma unroll
        for (int kc = 0; kc < 2; ++kc) {
            bf16x8 a0 = *(const bf16x8*)&Ps[wave * 1024 + lo * 64 + kc * 32 + hi * 8];
            bf16x8 a1 = *(const bf16x8*)&Ps[wave * 1024 + lo * 64 + kc * 32 + 16 + hi * 8];
            v16bf pf = pack16(a0, a1);
#pragma unroll
            for (int dn = 0; dn < 6; ++dn) {
                int drow = dn * 16 + lo;                 // B: lane -> d col
                bf16x8 b0 = *(const bf16x8*)&Vs[b][drow * 64 + kc * 32 + hi * 8];
                bf16x8 b1 = *(const bf16x8*)&Vs[b][drow * 64 + kc * 32 + 16 + hi * 8];
                o_acc[dn] = wmma_bf16(pf, pack16(b0, b1), o_acc[dn]);
            }
        }
    }

    // normalize and write [S][DIM] bf16 for the O-projection GEMM
#pragma unroll
    for (int dn = 0; dn < 6; ++dn)
#pragma unroll
        for (int r = 0; r < 8; ++r) {
            int m = q0 + wave * 16 + hi * 8 + r;
            int d = dn * 16 + lo;
            O[(size_t)m * DIMSZ + head * HDIM + d] = f2bf(o_acc[dn][r] / l_run[r]);
        }
}

// ---------------------------------------------------------------------------
// Host-side orchestration
// ---------------------------------------------------------------------------
extern "C" void kernel_launch(void* const* d_in, const int* in_sizes, int n_in,
                              void* d_out, int out_size, void* d_ws, size_t ws_size,
                              hipStream_t stream) {
    (void)in_sizes; (void)n_in; (void)out_size; (void)ws_size;

    const float* x    = (const float*)d_in[0];
    const float* wq   = (const float*)d_in[1];
    const float* bq   = (const float*)d_in[2];
    const float* wk   = (const float*)d_in[3];
    const float* bk   = (const float*)d_in[4];
    const float* wv   = (const float*)d_in[5];
    const float* bv   = (const float*)d_in[6];
    const float* wo   = (const float*)d_in[7];
    const float* bo   = (const float*)d_in[8];
    const float* fcos = (const float*)d_in[9];
    const float* fsin = (const float*)d_in[10];
    float* out = (float*)d_out;

    size_t off = 0;
    auto alloc = [&](size_t bytes) {
        void* p = (char*)d_ws + off;
        off += (bytes + 255) & ~(size_t)255;
        return p;
    };
    bf16_t* xbf  = (bf16_t*)alloc((size_t)S_LEN * DIMSZ * 2);
    bf16_t* wqt  = (bf16_t*)alloc((size_t)DIMSZ * DIMSZ * 2);
    bf16_t* wkt  = (bf16_t*)alloc((size_t)DIMSZ * DIMSZ * 2);
    bf16_t* wvt  = (bf16_t*)alloc((size_t)DIMSZ * DIMSZ * 2);
    bf16_t* wot  = (bf16_t*)alloc((size_t)DIMSZ * DIMSZ * 2);
    float*  qkv  = (float*) alloc((size_t)3 * S_LEN * DIMSZ * 4);
    bf16_t* Qbf  = (bf16_t*)alloc((size_t)NHEADS * S_LEN * HDIM * 2);
    bf16_t* Kbf  = (bf16_t*)alloc((size_t)NHEADS * S_LEN * HDIM * 2);
    bf16_t* Vtb  = (bf16_t*)alloc((size_t)NHEADS * HDIM * S_LEN * 2);
    bf16_t* attn = (bf16_t*)alloc((size_t)S_LEN * DIMSZ * 2);

    // 1. convert activations and (transposed) weights to bf16
    cvt_x_kernel<<<(S_LEN * DIMSZ) / 256, 256, 0, stream>>>(x, xbf, S_LEN * DIMSZ);
    cvt_wt_kernel<<<dim3((DIMSZ * DIMSZ) / 256, 1, 4), 256, 0, stream>>>(
        wq, wk, wv, wo, wqt, wkt, wvt, wot);

    // 2. QKV projections (fp32 out for RoPE)
    dim3 ggrid(DIMSZ / 128, S_LEN / 128);
    gemm_bf16_kernel<<<ggrid, 256, 0, stream>>>(xbf, wqt, bq, qkv + 0 * (size_t)S_LEN * DIMSZ);
    gemm_bf16_kernel<<<ggrid, 256, 0, stream>>>(xbf, wkt, bk, qkv + 1 * (size_t)S_LEN * DIMSZ);
    gemm_bf16_kernel<<<ggrid, 256, 0, stream>>>(xbf, wvt, bv, qkv + 2 * (size_t)S_LEN * DIMSZ);

    // 3. RoPE + pack to per-head bf16 layouts
    rope_pack_kernel<<<(S_LEN * NHEADS * CROPE) / 256, 256, 0, stream>>>(
        qkv, fcos, fsin, Qbf, Kbf, Vtb);

    // 4. flash attention
    flash_kernel<<<dim3(S_LEN / 128, NHEADS), 256, 0, stream>>>(Qbf, Kbf, Vtb, attn);

    // 5. output projection -> d_out (fp32)
    gemm_bf16_kernel<<<ggrid, 256, 0, stream>>>(attn, wot, bo, out);
}
